// FullAttentionBlock_60112362275432
// MI455X (gfx1250) — matrix operand order repaired and checked by reading, not compiled, not run on hardware
//
#include <hip/hip_runtime.h>
#include <hip/hip_bf16.h>

typedef _Float16 f16;
typedef __attribute__((ext_vector_type(16))) _Float16 v16h;
typedef __attribute__((ext_vector_type(8)))  _Float16 v8h;
typedef __attribute__((ext_vector_type(8)))  float    v8f;

#define B_    2
#define C_    512
#define N_    2304          // 48*48 tokens
#define NH_   8
#define DH_   64
#define HID_  512
#define NK_   2308          // N_ + 4 memory kv
#define NKP_  2336          // padded to 73 * 32
#define KIT_  73            // key iterations of 32
#define G_    32
#define CPG_  16            // channels per group
#define EPS_  1e-5f
#define SCALE_ 0.125f       // DH^-0.5
#define NEGINF_ (-1e30f)

// ---------------------------------------------------------------- WMMA helpers
__device__ __forceinline__ v8f wmma_f16(v16h a, v16h b, v8f c) {
    // D = A(16x32 f16) x B(32x16 f16) + C(16x16 f32)
    return __builtin_amdgcn_wmma_f32_16x16x32_f16(false, a, false, b,
                                                  (short)0, c, false, false);
}

// A-matrix 16x32 f16 fragment from row-major (M x K), row stride ld elements.
// ISA layout: lane m(0-15): K {0..7,16..23}; lane m+16: K {8..15,24..31}.
__device__ __forceinline__ v16h load_a(const f16* p, int ld) {
    int lane = threadIdx.x & 31;
    int row  = lane & 15;
    int kb   = (lane >> 4) * 8;
    const f16* r = p + (size_t)row * ld + kb;
    v8h lo = *(const v8h*)(r);
    v8h hi = *(const v8h*)(r + 16);
    v16h a;
#pragma unroll
    for (int i = 0; i < 8; ++i) { a[i] = lo[i]; a[i + 8] = hi[i]; }
    return a;
}

// B-matrix 32x16 f16 fragment from row-major (N x K) operand (i.e. GEMM with
// B transposed), row stride ld. ISA layout: lane n(0-15): K {0..7,16..23};
// lane n+16: K {8..15,24..31}.
__device__ __forceinline__ v16h load_b(const f16* p, int ld) {
    int lane = threadIdx.x & 31;
    int n    = lane & 15;
    int kb   = (lane >> 4) * 8;
    const f16* r = p + (size_t)n * ld + kb;
    v8h lo = *(const v8h*)(r);
    v8h hi = *(const v8h*)(r + 16);
    v16h b;
#pragma unroll
    for (int i = 0; i < 8; ++i) { b[i] = lo[i]; b[i + 8] = hi[i]; }
    return b;
}

// ---------------------------------------------------------------- async copy
// GLOBAL_LOAD_ASYNC_TO_LDS_B128: memory -> LDS DMA, tracked by ASYNCcnt.
// GVS addressing: 64-bit SGPR base + 32-bit per-lane byte offset.
// The LDS destination is the low 32 bits of the generic shared-pointer
__device__ __forceinline__ void async_copy_b128(void* lds_dst,
                                                const void* gbase,
                                                unsigned byte_off) {
    unsigned l = (unsigned)(uintptr_t)lds_dst;
    asm volatile("global_load_async_to_lds_b128 %0, %1, %2"
                 :: "v"(l), "v"(byte_off), "s"(gbase)
                 : "memory");
}
__device__ __forceinline__ void async_wait0() {
#if __has_builtin(__builtin_amdgcn_s_wait_asynccnt)
    __builtin_amdgcn_s_wait_asynccnt(0);
#else
    asm volatile("s_wait_asynccnt 0x0" ::: "memory");
#endif
}

// ---------------------------------------------------------------- GroupNorm 1
// x is (B,C,N): group slice is a contiguous run of 16*2304 floats.
__global__ void gn_stats_chw(const float* __restrict__ x, float* __restrict__ st) {
    __shared__ float ss[256], sq[256];
    int bg = blockIdx.x;                       // b*32 + g
    const float* p = x + (size_t)bg * (CPG_ * N_);
    float s = 0.f, q = 0.f;
    for (int i = threadIdx.x; i < CPG_ * N_; i += 256) {
        float v = p[i]; s += v; q += v * v;
    }
    ss[threadIdx.x] = s; sq[threadIdx.x] = q;
    __syncthreads();
    for (int d = 128; d; d >>= 1) {
        if ((int)threadIdx.x < d) { ss[threadIdx.x] += ss[threadIdx.x + d];
                                    sq[threadIdx.x] += sq[threadIdx.x + d]; }
        __syncthreads();
    }
    if (threadIdx.x == 0) {
        float inv = 1.f / (float)(CPG_ * N_);
        float m = ss[0] * inv;
        float var = sq[0] * inv - m * m;
        st[bg * 2 + 0] = m;
        st[bg * 2 + 1] = rsqrtf(var + EPS_);
    }
}

// apply GN1, transpose (B,C,N)->(B,N,C), cast f16
__global__ void gn1_apply(const float* __restrict__ x,
                          const float* __restrict__ gw, const float* __restrict__ gb,
                          const float* __restrict__ st, f16* __restrict__ t) {
    int gid = blockIdx.x * 256 + threadIdx.x;
    if (gid >= B_ * C_ * N_) return;
    int b = gid / (C_ * N_);
    int r = gid - b * (C_ * N_);
    int c = r / N_;
    int n = r - c * N_;
    int g = c >> 4;
    float mu = st[(b * G_ + g) * 2 + 0];
    float rs = st[(b * G_ + g) * 2 + 1];
    float v  = (x[gid] - mu) * rs * gw[c] + gb[c];
    t[((size_t)b * N_ + n) * C_ + c] = (f16)v;
}

__global__ void cast_f16(const float* __restrict__ s, f16* __restrict__ d, int n) {
    int i = blockIdx.x * 256 + threadIdx.x;
    if (i < n) d[i] = (f16)s[i];
}

// memory kv rows (0..3) + zero tail pad rows (2308..2335) of padded K/V buffers
__global__ void memkv_fill(const float* __restrict__ mkv,
                           f16* __restrict__ kh, f16* __restrict__ vh) {
    int gid = blockIdx.x * 256 + threadIdx.x;     // 2*2*8*32*64 = 65536
    int d   = gid & 63;
    int r5  = (gid >> 6) & 31;
    int h   = (gid >> 11) & 7;
    int b   = (gid >> 14) & 1;
    int seg = gid >> 15;
    f16* dst = seg ? vh : kh;
    int row; float val;
    if (r5 < 4) {   // mem_kv: (2, nh, dh, 4) -> row j of padded buffer
        row = r5;
        val = mkv[(((size_t)seg * NH_ + h) * DH_ + d) * 4 + r5];
    } else {        // zero padding
        row = NK_ + (r5 - 4);
        val = 0.f;
    }
    dst[(((size_t)b * NH_ + h) * NKP_ + row) * DH_ + d] = (f16)val;
}

// ---------------------------------------------------------------- QKV GEMM
// (4608 x 512) x (1536 x 512)^T, 8 waves/block, wave tile 16x64,
// weight chunk async-DMAed into LDS and shared across waves.
__global__ void qkv_gemm(const f16* __restrict__ T, const f16* __restrict__ Wq,
                         f16* __restrict__ qh, f16* __restrict__ kh,
                         f16* __restrict__ vh) {
    __shared__ alignas(16) f16 Wsh[64][32];
    int w    = threadIdx.x >> 5;
    int lane = threadIdx.x & 31;
    int r0   = blockIdx.x * 128 + w * 16;     // token-row tile
    int c0   = blockIdx.y * 64;               // output-col tile
    int mr   = threadIdx.x >> 2, ko = (threadIdx.x & 3) * 8;
    v8f acc[4] = {};
    for (int kc = 0; kc < C_; kc += 32) {
        __syncthreads();
        async_copy_b128(&Wsh[mr][ko], Wq,
                        (unsigned)(((c0 + mr) * C_ + kc + ko) * 2));
        async_wait0();
        __syncthreads();
        v16h a = load_a(T + (size_t)r0 * C_ + kc, C_);
#pragma unroll
        for (int ot = 0; ot < 4; ++ot) {
            v16h b = load_b(&Wsh[ot * 16][0], 32);
            acc[ot] = wmma_f16(a, b, acc[ot]);
        }
    }
    int half = lane >> 4, cn = lane & 15;
    int seg  = blockIdx.y >> 3;               // 0=q 1=k 2=v (64-col tiles)
#pragma unroll
    for (int ot = 0; ot < 4; ++ot) {
        int m = c0 + ot * 16 + cn;
        int h = (m >> 6) & 7, d = m & 63;
#pragma unroll
        for (int i = 0; i < 8; ++i) {
            int tok = r0 + i + 8 * half;
            int b   = (tok >= N_) ? 1 : 0;
            int n   = tok - b * N_;
            float v = acc[ot][i];
            if (seg == 0)
                qh[(((size_t)b * NH_ + h) * N_ + n) * DH_ + d] = (f16)(v * SCALE_);
            else if (seg == 1)
                kh[(((size_t)b * NH_ + h) * NKP_ + 4 + n) * DH_ + d] = (f16)v;
            else
                vh[(((size_t)b * NH_ + h) * NKP_ + 4 + n) * DH_ + d] = (f16)v;
        }
    }
}

// ---------------------------------------------------------------- attention
// block: 8 waves x 16 query rows = 128 rows per (b,h). Online softmax.
// K tile staged by async DMA (memory->LDS); V tile transposed via VGPRs.
__global__ void attn_fused(const f16* __restrict__ Q, const f16* __restrict__ K,
                           const f16* __restrict__ V, f16* __restrict__ O) {
    __shared__ alignas(16) f16 Ksh[32][64];     // 32 keys x 64 d
    __shared__ alignas(16) f16 Vt[64][32];      // transposed: d x key
    __shared__ alignas(16) f16 Pls[8][16][32];  // per-wave P bounce
    int bh   = blockIdx.x;                      // b*8 + h
    int w    = threadIdx.x >> 5;
    int lane = threadIdx.x & 31;
    int half = lane >> 4, cn = lane & 15;
    int q0   = blockIdx.y * 128 + w * 16;

    const f16* qp = Q + ((size_t)bh * N_ + q0) * DH_;
    v16h qa0 = load_a(qp, DH_);            // K dims 0..31  (q pre-scaled)
    v16h qa1 = load_a(qp + 32, DH_);       // K dims 32..63

    v8f o[4] = {};
    float mrow[8], lrow[8];
#pragma unroll
    for (int i = 0; i < 8; ++i) { mrow[i] = NEGINF_; lrow[i] = 0.f; }

    const f16* kbase = K + (size_t)bh * NKP_ * DH_;
    const f16* vbase = V + (size_t)bh * NKP_ * DH_;
    int r = threadIdx.x >> 3, dof = (threadIdx.x & 7) * 8;

    for (int kt = 0; kt < KIT_; ++kt) {
        int kb0 = kt * 32;
        __syncthreads();
        // K tile: async DMA straight into LDS (ASYNCcnt path)
        async_copy_b128(&Ksh[r][dof], kbase,
                        (unsigned)(((kb0 + r) * DH_ + dof) * 2));
        {   // V tile: VGPR round-trip for the transpose
            v8h vv = *(const v8h*)(vbase + (size_t)(kb0 + r) * DH_ + dof);
#pragma unroll
            for (int i = 0; i < 8; ++i) Vt[dof + i][r] = vv[i];
            if (kt + 1 < KIT_) {  // prefetch next tiles into cache
                __builtin_prefetch(kbase + (size_t)(kb0 + 32 + r) * DH_ + dof, 0, 0);
                __builtin_prefetch(vbase + (size_t)(kb0 + 32 + r) * DH_ + dof, 0, 0);
            }
        }
        async_wait0();
        __syncthreads();

        // S = Q x K^T  (two 16-key tiles, K=64 via two k=32 WMMA each)
        v8f s0 = {}, s1 = {};
        s0 = wmma_f16(qa0, load_b(&Ksh[0][0], 64), s0);
        s0 = wmma_f16(qa1, load_b(&Ksh[0][32], 64), s0);
        s1 = wmma_f16(qa0, load_b(&Ksh[16][0], 64), s1);
        s1 = wmma_f16(qa1, load_b(&Ksh[16][32], 64), s1);

        bool v0 = (kb0 + cn)      < NK_;
        bool v1 = (kb0 + 16 + cn) < NK_;
#pragma unroll
        for (int i = 0; i < 8; ++i) {
            float f0 = v0 ? s0[i] : NEGINF_;
            float f1 = v1 ? s1[i] : NEGINF_;
            float mx = fmaxf(f0, f1);
            for (int d = 8; d; d >>= 1) mx = fmaxf(mx, __shfl_xor(mx, d));
            float nm = fmaxf(mrow[i], mx);
            float al = __expf(mrow[i] - nm);
            float p0 = __expf(f0 - nm);
            float p1 = __expf(f1 - nm);
            float sm = p0 + p1;
            for (int d = 8; d; d >>= 1) sm += __shfl_xor(sm, d);
            lrow[i] = lrow[i] * al + sm;
            mrow[i] = nm;
#pragma unroll
            for (int ot = 0; ot < 4; ++ot) o[ot][i] *= al;
            int row = i + 8 * half;
            Pls[w][row][cn]      = (f16)p0;
            Pls[w][row][cn + 16] = (f16)p1;
        }

        // O += P(16x32) x V(32x64)
        v16h pa = load_a(&Pls[w][0][0], 32);
#pragma unroll
        for (int ot = 0; ot < 4; ++ot) {
            v16h vb = load_b(&Vt[ot * 16][0], 32);
            o[ot] = wmma_f16(pa, vb, o[ot]);
        }
    }

    // epilogue: normalize and store (B, N, hidden) f16
    int b = bh >> 3, h = bh & 7;
#pragma unroll
    for (int i = 0; i < 8; ++i) {
        float inv = 1.f / lrow[i];
        int row = i + 8 * half;
        int tok = q0 + row;
#pragma unroll
        for (int ot = 0; ot < 4; ++ot) {
            int d = ot * 16 + cn;
            O[((size_t)b * N_ + tok) * HID_ + h * DH_ + d] = (f16)(o[ot][i] * inv);
        }
    }
}

// ---------------------------------------------------------------- out proj
__global__ void out_gemm(const f16* __restrict__ A, const f16* __restrict__ Wo,
                         const float* __restrict__ bo, float* __restrict__ Y) {
    __shared__ alignas(16) f16 Wsh[64][32];
    int w    = threadIdx.x >> 5;
    int lane = threadIdx.x & 31;
    int r0   = blockIdx.x * 128 + w * 16;
    int c0   = blockIdx.y * 64;
    int mr   = threadIdx.x >> 2, ko = (threadIdx.x & 3) * 8;
    v8f acc[4] = {};
    for (int kc = 0; kc < HID_; kc += 32) {
        __syncthreads();
        async_copy_b128(&Wsh[mr][ko], Wo,
                        (unsigned)(((c0 + mr) * HID_ + kc + ko) * 2));
        async_wait0();
        __syncthreads();
        v16h a = load_a(A + (size_t)r0 * HID_ + kc, HID_);
#pragma unroll
        for (int ot = 0; ot < 4; ++ot) {
            v16h b = load_b(&Wsh[ot * 16][0], 32);
            acc[ot] = wmma_f16(a, b, acc[ot]);
        }
    }
    int half = lane >> 4, cn = lane & 15;
#pragma unroll
    for (int ot = 0; ot < 4; ++ot) {
        int m = c0 + ot * 16 + cn;
#pragma unroll
        for (int i = 0; i < 8; ++i) {
            int tok = r0 + i + 8 * half;
            Y[(size_t)tok * C_ + m] = acc[ot][i] + bo[m];
        }
    }
}

// ---------------------------------------------------------------- GroupNorm 2
__global__ void gn_stats_nc(const float* __restrict__ y, float* __restrict__ st) {
    __shared__ float ss[256], sq[256];
    int bg = blockIdx.x;
    int b = bg >> 5, g = bg & 31;
    const float* p = y + (size_t)b * N_ * C_ + g * CPG_;
    float s = 0.f, q = 0.f;
    for (int i = threadIdx.x; i < CPG_ * N_; i += 256) {
        int n = i >> 4, c = i & 15;
        float v = p[(size_t)n * C_ + c];
        s += v; q += v * v;
    }
    ss[threadIdx.x] = s; sq[threadIdx.x] = q;
    __syncthreads();
    for (int d = 128; d; d >>= 1) {
        if ((int)threadIdx.x < d) { ss[threadIdx.x] += ss[threadIdx.x + d];
                                    sq[threadIdx.x] += sq[threadIdx.x + d]; }
        __syncthreads();
    }
    if (threadIdx.x == 0) {
        float inv = 1.f / (float)(CPG_ * N_);
        float m = ss[0] * inv;
        float var = sq[0] * inv - m * m;
        st[bg * 2 + 0] = m;
        st[bg * 2 + 1] = rsqrtf(var + EPS_);
    }
}

// apply GN2 + transpose (B,N,C) -> (B,C,H,W) f32 output
__global__ void gn2_apply(const float* __restrict__ y,
                          const float* __restrict__ gw, const float* __restrict__ gb,
                          const float* __restrict__ st, float* __restrict__ out) {
    int gid = blockIdx.x * 256 + threadIdx.x;
    if (gid >= B_ * C_ * N_) return;
    int b = gid / (C_ * N_);
    int r = gid - b * (C_ * N_);
    int c = r / N_;
    int n = r - c * N_;
    int g = c >> 4;
    float mu = st[(b * G_ + g) * 2 + 0];
    float rs = st[(b * G_ + g) * 2 + 1];
    float v  = y[((size_t)b * N_ + n) * C_ + c];
    out[gid] = (v - mu) * rs * gw[c] + gb[c];
}

// ---------------------------------------------------------------- launch
extern "C" void kernel_launch(void* const* d_in, const int* in_sizes, int n_in,
                              void* d_out, int out_size, void* d_ws, size_t ws_size,
                              hipStream_t stream) {
    const float* x     = (const float*)d_in[0];
    const float* gn1_w = (const float*)d_in[1];
    const float* gn1_b = (const float*)d_in[2];
    const float* w_qkv = (const float*)d_in[3];
    const float* memkv = (const float*)d_in[4];
    const float* w_out = (const float*)d_in[5];
    const float* b_out = (const float*)d_in[6];
    const float* gn2_w = (const float*)d_in[7];
    const float* gn2_b = (const float*)d_in[8];
    float* out = (float*)d_out;

    // workspace carve-up (256B aligned)
    size_t off = 0;
    auto carve = [&](size_t bytes) {
        size_t o = off; off += (bytes + 255) & ~(size_t)255; return o;
    };
    char* ws = (char*)d_ws;
    f16*   tf16  = (f16*)(ws + carve((size_t)B_ * N_ * C_ * 2));
    f16*   wqkvh = (f16*)(ws + carve((size_t)3 * HID_ * C_ * 2));
    f16*   wouth = (f16*)(ws + carve((size_t)C_ * HID_ * 2));
    f16*   qh    = (f16*)(ws + carve((size_t)B_ * NH_ * N_ * DH_ * 2));
    f16*   kh    = (f16*)(ws + carve((size_t)B_ * NH_ * NKP_ * DH_ * 2));
    f16*   vh    = (f16*)(ws + carve((size_t)B_ * NH_ * NKP_ * DH_ * 2));
    f16*   oh    = (f16*)(ws + carve((size_t)B_ * N_ * HID_ * 2));
    float* y     = (float*)(ws + carve((size_t)B_ * N_ * C_ * 4));
    float* st1   = (float*)(ws + carve((size_t)B_ * G_ * 2 * 4));
    float* st2   = (float*)(ws + carve((size_t)B_ * G_ * 2 * 4));
    (void)in_sizes; (void)n_in; (void)out_size; (void)ws_size;

    const int total = B_ * C_ * N_;

    gn_stats_chw<<<B_ * G_, 256, 0, stream>>>(x, st1);
    gn1_apply<<<(total + 255) / 256, 256, 0, stream>>>(x, gn1_w, gn1_b, st1, tf16);
    cast_f16<<<(3 * HID_ * C_ + 255) / 256, 256, 0, stream>>>(w_qkv, wqkvh, 3 * HID_ * C_);
    cast_f16<<<(C_ * HID_ + 255) / 256, 256, 0, stream>>>(w_out, wouth, C_ * HID_);
    memkv_fill<<<(2 * B_ * NH_ * 32 * DH_) / 256, 256, 0, stream>>>(memkv, kh, vh);

    qkv_gemm<<<dim3((B_ * N_) / 128, (3 * HID_) / 64), 256, 0, stream>>>(
        tf16, wqkvh, qh, kh, vh);
    attn_fused<<<dim3(B_ * NH_, N_ / 128), 256, 0, stream>>>(qh, kh, vh, oh);
    out_gemm<<<dim3((B_ * N_) / 128, C_ / 64), 256, 0, stream>>>(oh, wouth, b_out, y);

    gn_stats_nc<<<B_ * G_, 256, 0, stream>>>(y, st2);
    gn2_apply<<<(total + 255) / 256, 256, 0, stream>>>(y, gn2_w, gn2_b, st2, out);
}
